// HAN_62921270886522
// MI455X (gfx1250) — matrix operand order repaired and checked by reading, not compiled, not run on hardware
//
#include <hip/hip_runtime.h>
#include <hip/hip_bf16.h>
#include <math.h>

// ---------------------------------------------------------------------------
// HAN layer for MI455X (gfx1250): bf16 WMMA GEMMs + atomic segment softmax
// ---------------------------------------------------------------------------

#define HIDDIM 128
#define NHEAD  8
#define DHEAD  16

typedef __attribute__((ext_vector_type(16))) __bf16 v16bf;
typedef __attribute__((ext_vector_type(8)))  float  v8f;

__device__ __forceinline__ unsigned short f2bf(float f) {
    unsigned u = __float_as_uint(f);
    u += 0x7FFFu + ((u >> 16) & 1u);          // round-to-nearest-even
    return (unsigned short)(u >> 16);
}
__device__ __forceinline__ unsigned pack2(float a, float b) {
    return (unsigned)f2bf(a) | ((unsigned)f2bf(b) << 16);
}
// order-preserving float <-> uint mapping so atomicMax(uint) == float max
__device__ __forceinline__ unsigned enc_f(float f) {
    unsigned u = __float_as_uint(f);
    return (u >> 31) ? ~u : (u | 0x80000000u);
}
__device__ __forceinline__ float dec_f(unsigned eu) {
    return (eu >> 31) ? __uint_as_float(eu & 0x7FFFFFFFu) : __uint_as_float(~eu);
}

// ---------------------------------------------------------------------------
// WMMA GEMM: OUT[M x 128] = X[M x 128] @ W[128 x 128]^T (+ bias)
// SCORE==1: instead of storing, accumulate sum_n tanh(out)*q into *score_out
// block = 256 threads (8 waves); each wave computes a 16-row x 128-col strip
// ---------------------------------------------------------------------------
template <int SCORE>
__global__ __launch_bounds__(256) void gemm128_wmma(
    const float* __restrict__ X, const float* __restrict__ W,
    const float* __restrict__ bias, const float* __restrict__ qv,
    float* __restrict__ OUT, float* __restrict__ score_out, int M)
{
    __shared__ __align__(16) unsigned short sW[128 * 132];  // W as bf16, padded
    __shared__ __align__(16) unsigned short sA[128 * 132];  // X tile as bf16

    const int tid = threadIdx.x;
    const int rowBase = blockIdx.x * 128;

    // stage W (row-major [n][k]) into LDS as bf16
    for (int i = tid; i < 4096; i += 256) {
        int r = i >> 5, c4 = i & 31;
        float4 wv = ((const float4*)(W + r * 128))[c4];
        *(uint2*)&sW[r * 132 + c4 * 4] =
            make_uint2(pack2(wv.x, wv.y), pack2(wv.z, wv.w));
    }
    // stage 128 rows of X into LDS as bf16 (zeros past M)
    for (int i = tid; i < 4096; i += 256) {
        int r = i >> 5, c4 = i & 31;
        int grow = rowBase + r;
        uint2 p = make_uint2(0u, 0u);
        if (grow < M) {
            float4 xv = ((const float4*)(X + (size_t)grow * 128))[c4];
            p = make_uint2(pack2(xv.x, xv.y), pack2(xv.z, xv.w));
        }
        *(uint2*)&sA[r * 132 + c4 * 4] = p;
    }
    __syncthreads();

    const int wave  = tid >> 5;
    const int lane  = tid & 31;
    const int lm    = lane & 15;
    const int lhalf = lane >> 4;
    const int am    = (wave << 4) + lm;   // A row within block tile

    v8f acc[8];
#pragma unroll
    for (int t = 0; t < 8; ++t)
#pragma unroll
        for (int j = 0; j < 8; ++j) acc[t][j] = 0.0f;

#pragma unroll
    for (int kk = 0; kk < 4; ++kk) {              // K = 4 x 32
        // A fragment: 16x32 bf16 per documented gfx1250 layout
        union { v16bf v; unsigned u[8]; } af;
        const int kA = kk * 32 + lhalf * 8;
#pragma unroll
        for (int v = 0; v < 8; ++v) {
            int k = kA + ((v >> 2) << 4) + ((v & 3) << 1);
            af.u[v] = *(const unsigned*)&sA[am * 132 + k];
        }
#pragma unroll
        for (int nt = 0; nt < 8; ++nt) {          // 8 column tiles of 16
            union { v16bf v; unsigned u[8]; } bf;
            const int bn = nt * 16 + lm;          // B col n -> W row n
            const int kB = kk * 32 + lhalf * 8;
#pragma unroll
            for (int v = 0; v < 8; ++v)
                bf.u[v] = *(const unsigned*)&sW[bn * 132 + kB + ((v >> 2) << 4) + ((v & 3) << 1)];
            acc[nt] = __builtin_amdgcn_wmma_f32_16x16x32_bf16(
                false, af.v, false, bf.v, (short)0, acc[nt], false, false);
        }
    }

    const int gm0 = rowBase + (wave << 4) + lhalf * 8;  // C: row = 8*half + v
    if (SCORE == 0) {
#pragma unroll
        for (int nt = 0; nt < 8; ++nt) {
            int col = nt * 16 + lm;
            float b = bias[col];
#pragma unroll
            for (int v = 0; v < 8; ++v) {
                int grow = gm0 + v;
                if (grow < M) OUT[(size_t)grow * 128 + col] = acc[nt][v] + b;
            }
        }
    } else {
        float partial = 0.0f;
#pragma unroll
        for (int nt = 0; nt < 8; ++nt) {
            int col = nt * 16 + lm;
            float b  = bias[col];
            float qc = qv[col];
#pragma unroll
            for (int v = 0; v < 8; ++v) {
                int grow = gm0 + v;
                if (grow < M) partial += tanhf(acc[nt][v] + b) * qc;
            }
        }
#pragma unroll
        for (int o = 16; o >= 1; o >>= 1)
            partial += __shfl_xor(partial, o, 32);   // wave32 reduce
        if (lane == 0) atomicAdd(score_out, partial);
    }
}

// ---------------------------------------------------------------------------
// per-head attention logits: out[n,h] = sum_d x[n, h*16+d] * att[h, d]
// ---------------------------------------------------------------------------
__global__ void att_dot_kernel(const float* __restrict__ x,
                               const float* __restrict__ att,
                               float* __restrict__ out, int N)
{
    int t = blockIdx.x * blockDim.x + threadIdx.x;
    if (t >= N * NHEAD) return;
    int n = t >> 3, h = t & 7;
    const float* xv = x + (size_t)n * 128 + h * DHEAD;
    const float* av = att + h * DHEAD;
    float s = 0.f;
#pragma unroll
    for (int d = 0; d < DHEAD; ++d) s += xv[d] * av[d];
    out[t] = s;
}

// ---------------------------------------------------------------------------
// Pass A: segment max over dst (float max via ordered-uint atomicMax)
// ---------------------------------------------------------------------------
__global__ void edge_max_kernel(const int* __restrict__ src,
                                const int* __restrict__ dst,
                                const float* __restrict__ asrc,
                                const float* __restrict__ adst,
                                unsigned* __restrict__ mbuf, int E)
{
    int t = blockIdx.x * blockDim.x + threadIdx.x;
    if (t >= E * NHEAD) return;
    int e = t >> 3, h = t & 7;
    int s = src[e], d = dst[e];
    float a = asrc[s * NHEAD + h] + adst[d * NHEAD + h];
    a = a > 0.f ? a : 0.2f * a;                    // leaky_relu(0.2)
    atomicMax(&mbuf[d * NHEAD + h], enc_f(a));
}

// ---------------------------------------------------------------------------
// Pass B: segment sum of exp(a - m)
// ---------------------------------------------------------------------------
__global__ void edge_sum_kernel(const int* __restrict__ src,
                                const int* __restrict__ dst,
                                const float* __restrict__ asrc,
                                const float* __restrict__ adst,
                                const unsigned* __restrict__ mbuf,
                                float* __restrict__ sbuf, int E)
{
    int t = blockIdx.x * blockDim.x + threadIdx.x;
    if (t >= E * NHEAD) return;
    int e = t >> 3, h = t & 7;
    int s = src[e], d = dst[e];
    float a = asrc[s * NHEAD + h] + adst[d * NHEAD + h];
    a = a > 0.f ? a : 0.2f * a;
    float m = dec_f(mbuf[d * NHEAD + h]);
    if (!(fabsf(m) <= 3.0e38f)) m = 0.f;           // isfinite guard
    atomicAdd(&sbuf[d * NHEAD + h], expf(a - m));
}

// ---------------------------------------------------------------------------
// Pass C: one wave per edge; lanes 0..7 compute alpha[h], broadcast via shfl,
// then 4 coalesced 32-lane sweeps of atomicAdd(out[dst, f], x[src, f]*alpha)
// ---------------------------------------------------------------------------
__global__ __launch_bounds__(256) void edge_msg_kernel(
    const int* __restrict__ src, const int* __restrict__ dst,
    const float* __restrict__ asrc, const float* __restrict__ adst,
    const unsigned* __restrict__ mbuf, const float* __restrict__ sbuf,
    const float* __restrict__ xsrc, float* __restrict__ out, int E)
{
    int wid  = (blockIdx.x * blockDim.x + threadIdx.x) >> 5;
    int lane = threadIdx.x & 31;
    if (wid >= E) return;
    int s = src[wid], d = dst[wid];

    float alpha = 0.f;
    if (lane < NHEAD) {
        float a = asrc[s * NHEAD + lane] + adst[d * NHEAD + lane];
        a = a > 0.f ? a : 0.2f * a;
        float m = dec_f(mbuf[d * NHEAD + lane]);
        if (!(fabsf(m) <= 3.0e38f)) m = 0.f;
        alpha = expf(a - m) / (sbuf[d * NHEAD + lane] + 1e-16f);
    }
    const float* xr = xsrc + (size_t)s * 128;
    float* orow = out + (size_t)d * 128;
#pragma unroll
    for (int i = 0; i < 4; ++i) {
        int f = i * 32 + lane;
        float al = __shfl(alpha, f >> 4, 32);
        atomicAdd(&orow[f], xr[f] * al);
    }
}

__global__ void relu_kernel(float* __restrict__ p, size_t n)
{
    size_t i = (size_t)blockIdx.x * blockDim.x + threadIdx.x;
    if (i < n) { float v = p[i]; p[i] = v > 0.f ? v : 0.f; }
}

__global__ void zero_kernel(float* __restrict__ p, size_t n)
{
    size_t i = (size_t)blockIdx.x * blockDim.x + threadIdx.x;
    if (i < n) p[i] = 0.0f;
}

__global__ void softmax2_kernel(const float* __restrict__ score_acc,
                                float* __restrict__ attn, float invN)
{
    float s0 = score_acc[0] * invN, s1 = score_acc[1] * invN;  // mean over nodes
    float mx = fmaxf(s0, s1);
    float e0 = expf(s0 - mx), e1 = expf(s1 - mx);
    float inv = 1.f / (e0 + e1);
    attn[0] = e0 * inv;
    attn[1] = e1 * inv;
}

// fused = attn0*out_gg + attn1*out_dg; out = fused @ lin_w.T + lin_b
__global__ __launch_bounds__(128) void final_kernel(
    const float* __restrict__ og, const float* __restrict__ od,
    const float* __restrict__ attn, const float* __restrict__ lw,
    const float* __restrict__ lb, float* __restrict__ out, int N)
{
    int n = blockIdx.x, t = threadIdx.x;
    float a0 = attn[0], a1 = attn[1];
    float f = a0 * og[(size_t)n * 128 + t] + a1 * od[(size_t)n * 128 + t];
    float p0 = f * lw[t];
    float p1 = f * lw[128 + t];
#pragma unroll
    for (int o = 16; o >= 1; o >>= 1) {
        p0 += __shfl_xor(p0, o, 32);
        p1 += __shfl_xor(p1, o, 32);
    }
    __shared__ float r0[4], r1[4];
    int w = t >> 5, l = t & 31;
    if (l == 0) { r0[w] = p0; r1[w] = p1; }
    __syncthreads();
    if (t == 0) {
        out[(size_t)n * 2 + 0] = r0[0] + r0[1] + r0[2] + r0[3] + lb[0];
        out[(size_t)n * 2 + 1] = r1[0] + r1[1] + r1[2] + r1[3] + lb[1];
    }
}

// ---------------------------------------------------------------------------
extern "C" void kernel_launch(void* const* d_in, const int* in_sizes, int n_in,
                              void* d_out, int out_size, void* d_ws, size_t ws_size,
                              hipStream_t stream)
{
    const float* x_gene    = (const float*)d_in[0];
    const float* x_disease = (const float*)d_in[1];
    const int*   edge_gg   = (const int*)d_in[2];
    const int*   edge_dg   = (const int*)d_in[3];
    const float* pg_w      = (const float*)d_in[4];
    const float* pg_b      = (const float*)d_in[5];
    const float* pd_w      = (const float*)d_in[6];
    const float* pd_b      = (const float*)d_in[7];
    const float* att_s_gg  = (const float*)d_in[8];
    const float* att_d_gg  = (const float*)d_in[9];
    const float* att_s_dg  = (const float*)d_in[10];
    const float* att_d_dg  = (const float*)d_in[11];
    const float* k_lin_w   = (const float*)d_in[12];
    const float* k_lin_b   = (const float*)d_in[13];
    const float* q         = (const float*)d_in[14];
    const float* lin_w     = (const float*)d_in[15];
    const float* lin_b     = (const float*)d_in[16];
    float*       out       = (float*)d_out;

    const int NG  = in_sizes[0] / 128;
    const int ND  = in_sizes[1] / 128;
    const int EGG = in_sizes[2] / 2;
    const int EDG = in_sizes[3] / 2;

    const int* src_gg = edge_gg;
    const int* dst_gg = edge_gg + EGG;
    const int* src_dg = edge_dg;
    const int* dst_dg = edge_dg + EDG;

    // workspace layout (4-byte elements)
    float* ws = (float*)d_ws;
    size_t o = 0;
    float* xg      = ws + o; o += (size_t)NG * 128;
    float* xd      = ws + o; o += (size_t)ND * 128;
    float* asrc_gg = ws + o; o += (size_t)NG * NHEAD;
    float* adst_gg = ws + o; o += (size_t)NG * NHEAD;
    float* adst_dg = ws + o; o += (size_t)NG * NHEAD;
    float* asrc_dg = ws + o; o += (size_t)ND * NHEAD;
    size_t zstart = o;                       // everything below gets zeroed
    float* out_gg  = ws + o; o += (size_t)NG * 128;
    float* out_dg  = ws + o; o += (size_t)NG * 128;
    unsigned* m_gg = (unsigned*)(ws + o); o += (size_t)NG * NHEAD;
    float*    s_gg = ws + o;              o += (size_t)NG * NHEAD;
    unsigned* m_dg = (unsigned*)(ws + o); o += (size_t)NG * NHEAD;
    float*    s_dg = ws + o;              o += (size_t)NG * NHEAD;
    float* score   = ws + o; o += 2;
    float* attn    = ws + o; o += 2;
    size_t zcount = o - zstart;
    (void)ws_size; (void)n_in; (void)out_size;

    // 0) zero accumulators (uint 0 == float 0 bit pattern == encoded -inf)
    zero_kernel<<<(unsigned)((zcount + 255) / 256), 256, 0, stream>>>(ws + zstart, zcount);

    // 1) projections via bf16 WMMA
    gemm128_wmma<0><<<(NG + 127) / 128, 256, 0, stream>>>(x_gene, pg_w, pg_b, nullptr, xg, nullptr, NG);
    gemm128_wmma<0><<<(ND + 127) / 128, 256, 0, stream>>>(x_disease, pd_w, pd_b, nullptr, xd, nullptr, ND);

    // 2) per-head attention logits
    att_dot_kernel<<<(NG * NHEAD + 255) / 256, 256, 0, stream>>>(xg, att_s_gg, asrc_gg, NG);
    att_dot_kernel<<<(NG * NHEAD + 255) / 256, 256, 0, stream>>>(xg, att_d_gg, adst_gg, NG);
    att_dot_kernel<<<(NG * NHEAD + 255) / 256, 256, 0, stream>>>(xg, att_d_dg, adst_dg, NG);
    att_dot_kernel<<<(ND * NHEAD + 255) / 256, 256, 0, stream>>>(xd, att_s_dg, asrc_dg, ND);

    // 3) segment softmax + message aggregation, per edge type
    edge_max_kernel<<<(EGG * NHEAD + 255) / 256, 256, 0, stream>>>(src_gg, dst_gg, asrc_gg, adst_gg, m_gg, EGG);
    edge_max_kernel<<<(EDG * NHEAD + 255) / 256, 256, 0, stream>>>(src_dg, dst_dg, asrc_dg, adst_dg, m_dg, EDG);
    edge_sum_kernel<<<(EGG * NHEAD + 255) / 256, 256, 0, stream>>>(src_gg, dst_gg, asrc_gg, adst_gg, m_gg, s_gg, EGG);
    edge_sum_kernel<<<(EDG * NHEAD + 255) / 256, 256, 0, stream>>>(src_dg, dst_dg, asrc_dg, adst_dg, m_dg, s_dg, EDG);
    edge_msg_kernel<<<(EGG + 7) / 8, 256, 0, stream>>>(src_gg, dst_gg, asrc_gg, adst_gg, m_gg, s_gg, xg, out_gg, EGG);
    edge_msg_kernel<<<(EDG + 7) / 8, 256, 0, stream>>>(src_dg, dst_dg, asrc_dg, adst_dg, m_dg, s_dg, xd, out_dg, EDG);

    // 4) relu on both metapath outputs (contiguous)
    {
        size_t n = (size_t)NG * 128 * 2;
        relu_kernel<<<(unsigned)((n + 255) / 256), 256, 0, stream>>>(out_gg, n);
    }

    // 5) semantic attention scores via fused WMMA + tanh + q-dot reduction
    gemm128_wmma<1><<<(NG + 127) / 128, 256, 0, stream>>>(out_gg, k_lin_w, k_lin_b, q, nullptr, score + 0, NG);
    gemm128_wmma<1><<<(NG + 127) / 128, 256, 0, stream>>>(out_dg, k_lin_w, k_lin_b, q, nullptr, score + 1, NG);
    softmax2_kernel<<<1, 1, 0, stream>>>(score, attn, 1.0f / (float)NG);

    // 6) fuse metapaths + final linear [128 -> 2]
    final_kernel<<<NG, 128, 0, stream>>>(out_gg, out_dg, attn, lin_w, lin_b, out, NG);
}
